// GraphSearchPolicy_38697655337446
// MI455X (gfx1250) — matrix operand order repaired and checked by reading, not compiled, not run on hardware
//
#include <hip/hip_runtime.h>
#include <hip/hip_bf16.h>

// ---------------------------------------------------------------------------
// Problem constants (match reference)
// ---------------------------------------------------------------------------
#define B_DIM   2048
#define K_DIM   256
#define ED      256
#define AD      512        // action_dim
#define IN_DIM  768        // ED + HD + RD
#define HUGE_F  1e31f

typedef __attribute__((ext_vector_type(16))) __bf16 v16bf;
typedef __attribute__((ext_vector_type(8)))  __bf16 bf16x8;
typedef __attribute__((ext_vector_type(4)))  __bf16 bf16x4;
typedef __attribute__((ext_vector_type(8)))  float  v8f;

// ---------------------------------------------------------------------------
// fp32 -> bf16 elementwise convert (weights; L2-resident, trivial)
// ---------------------------------------------------------------------------
__global__ void cvt_f32_bf16_kernel(const float* __restrict__ src,
                                    __bf16* __restrict__ dst, int n) {
    int i = blockIdx.x * blockDim.x + threadIdx.x;
    if (i < n) dst[i] = (__bf16)src[i];
}

// ---------------------------------------------------------------------------
// Pack Xin = bf16(concat([E, H, Q], -1))  ->  [B_DIM, IN_DIM]
// ---------------------------------------------------------------------------
__global__ __launch_bounds__(256) void
pack_xin_kernel(const float* __restrict__ E, const float* __restrict__ H,
                const float* __restrict__ Q, __bf16* __restrict__ Xin) {
    const int idx = blockIdx.x * blockDim.x + threadIdx.x;   // one float4
    const int b   = idx / (IN_DIM / 4);
    const int c4  = (idx % (IN_DIM / 4)) * 4;
    const float* src;
    if (c4 < 256)      src = E + b * 256 + c4;
    else if (c4 < 512) src = H + b * 256 + (c4 - 256);
    else               src = Q + b * 256 + (c4 - 512);
    float4 f = *(const float4*)src;
    bf16x4 v;
    v[0] = (__bf16)f.x; v[1] = (__bf16)f.y;
    v[2] = (__bf16)f.z; v[3] = (__bf16)f.w;
    *(bf16x4*)(Xin + (long)b * IN_DIM + c4) = v;
}

// ---------------------------------------------------------------------------
// Fragment helpers (layouts per CDNA5 ISA 7.12.2, wave32)
//   A 16x32 bf16 : lane l holds row (l&15), K = {kh*8..+7, kh*8+16..+23}
//   B 32x16 bf16 : lane l holds col (l&15), K = {kh*8..+7, kh*8+16..+23} (same as A)
//   C/D 16x16 f32: element v -> row v + 8*(l>>4), col (l&15)
// ---------------------------------------------------------------------------
__device__ __forceinline__ v16bf load_a_frag(const __bf16* arow_k0, int khalf) {
    const __bf16* p = arow_k0 + khalf * 8;
    bf16x8 lo = *(const bf16x8*)(p);
    bf16x8 hi = *(const bf16x8*)(p + 16);
    return __builtin_shufflevector(lo, hi, 0,1,2,3,4,5,6,7,8,9,10,11,12,13,14,15);
}

// ---------------------------------------------------------------------------
// CDNA5 async global->LDS copy (tracked on ASYNCcnt, not LOADcnt).
// lds_addr: LDS byte offset (addr[31:0] of the generic pointer, ISA 10.2).
// ---------------------------------------------------------------------------
__device__ __forceinline__ void async_load_b128(unsigned lds_addr,
                                                const void* gaddr) {
    asm volatile("global_load_async_to_lds_b128 %0, %1, off"
                 :: "v"(lds_addr), "v"((unsigned long long)(uintptr_t)gaddr)
                 : "memory");
}
__device__ __forceinline__ void wait_asynccnt0() {
    asm volatile("s_wait_asynccnt 0x0" ::: "memory");
}

// ---------------------------------------------------------------------------
// bf16 WMMA GEMM:  out = act(A[BxKD] @ W[ADxKD]^T + bias)
// Block (8 waves) -> 16 rows x 512 cols; wave w -> cols [w*64, w*64+64).
// Per K-step the block's 512x32 weight slab (32 KB) is staged into LDS with
// async-DMA double buffering: step s+1's slab streams in on ASYNCcnt while
// the WMMAs for step s consume ds_load_b128 fragments from the other buffer.
// ---------------------------------------------------------------------------
template <int KD, bool FUSED_RELU_BF16>
__global__ __launch_bounds__(256) void
gemm_wmma_kernel(const __bf16* __restrict__ A, const __bf16* __restrict__ W,
                 const float* __restrict__ bias,
                 __bf16* __restrict__ out_bf, float* __restrict__ out_f32) {
    __shared__ __bf16 wslab[2][AD * 32];          // 2 x 32 KB

    const int t     = threadIdx.x;                // 0..255
    const int lane  = t & 31;
    const int ng    = t >> 5;                     // wave id == n-group (0..7)
    const int mt    = blockIdx.x;                 // 0..127
    const int mbase = mt * 16;
    const int l15   = lane & 15;
    const int khalf = lane >> 4;
    const int row   = mbase + l15;
    const int NSTEP = KD / 32;

    const unsigned lds_base = (unsigned)(uintptr_t)&wslab[0][0];
    const unsigned slab_bytes = AD * 32 * 2;      // 32768

    const __bf16* arow = A + (long)row * KD;

    // issue the 32 KB slab for K-chunk [k0,k0+32) into buffer `buf`
    auto issue_slab = [&](int buf, int k0) {
        #pragma unroll
        for (int i = 0; i < 8; ++i) {
            const int c   = i * 256 + t;          // 16B chunk id, 0..2047
            const int r   = c >> 2;               // weight row 0..511
            const int seg = c & 3;                // 16B piece of 64B row-chunk
            const __bf16* g = W + (long)r * KD + k0 + seg * 8;
            async_load_b128(lds_base + (unsigned)buf * slab_bytes + c * 16, g);
        }
    };

    v8f acc[4] = {v8f{}, v8f{}, v8f{}, v8f{}};

    // prologue: stage slab 0, prefetch A fragment 0
    issue_slab(0, 0);
    v16bf a_cur = load_a_frag(arow, khalf);
    wait_asynccnt0();
    __syncthreads();

    for (int s = 0; s < NSTEP; ++s) {
        const int k0  = s * 32;
        const int buf = s & 1;
        v16bf a_nxt = a_cur;
        if (s + 1 < NSTEP) {
            issue_slab(buf ^ 1, k0 + 32);         // background DMA (ASYNCcnt)
            a_nxt = load_a_frag(arow + k0 + 32, khalf);
        }
        // consume current slab: ds_load_b128 fragments + 4 WMMAs
        #pragma unroll
        for (int j = 0; j < 4; ++j) {
            const int wrow = ng * 64 + j * 16 + l15;
            const __bf16* p = &wslab[buf][wrow * 32 + khalf * 8];
            bf16x8 lo = *(const bf16x8*)(p);
            bf16x8 hi = *(const bf16x8*)(p + 16);
            v16bf b = __builtin_shufflevector(lo, hi,
                          0,1,2,3,4,5,6,7,8,9,10,11,12,13,14,15);
            acc[j] = __builtin_amdgcn_wmma_f32_16x16x32_bf16(
                         false, a_cur, false, b, (short)0, acc[j],
                         false, false);
        }
        a_cur = a_nxt;
        wait_asynccnt0();                          // next slab landed
        __syncthreads();                           // all waves done reading
    }

    #pragma unroll
    for (int j = 0; j < 4; ++j) {
        const int nb   = ng * 64 + j * 16 + l15;
        const float bs = bias[nb];
        #pragma unroll
        for (int v = 0; v < 8; ++v) {
            const int rr = mbase + v + khalf * 8;
            float val = acc[j][v] + bs;
            if (FUSED_RELU_BF16) {
                val = val > 0.0f ? val : 0.0f;
                out_bf[(long)rr * AD + nb] = (__bf16)val;
            } else {
                out_f32[(long)rr * AD + nb] = val;
            }
        }
    }
}

// ---------------------------------------------------------------------------
// Scores + masked softmax + entropy.  One block (8 waves) per batch row;
// thread k owns action k.  Batched GEMV is L2-gather-bound (emb_e = 100 MB,
// resident in the 192 MB L2): VALU fp32 FMA, X2 row cached in LDS.
// ---------------------------------------------------------------------------
__device__ __forceinline__ float wave_max(float v) {
    #pragma unroll
    for (int off = 16; off > 0; off >>= 1)
        v = fmaxf(v, __shfl_xor(v, off, 32));
    return v;
}
__device__ __forceinline__ float wave_sum(float v) {
    #pragma unroll
    for (int off = 16; off > 0; off >>= 1)
        v += __shfl_xor(v, off, 32);
    return v;
}

__global__ __launch_bounds__(256) void
scores_kernel(const float* __restrict__ X2, const float* __restrict__ emb_r,
              const float* __restrict__ emb_e, const float* __restrict__ mask,
              const int* __restrict__ r_space, const int* __restrict__ e_space,
              float* __restrict__ dist, float* __restrict__ entropy) {
    __shared__ float xs[AD];
    __shared__ float red[8];

    const int b    = blockIdx.x;
    const int t    = threadIdx.x;          // 0..255 == action index
    const int lane = t & 31;
    const int wid  = t >> 5;

    xs[t]       = X2[b * AD + t];
    xs[256 + t] = X2[b * AD + 256 + t];
    __syncthreads();

    const int r = r_space[b * K_DIM + t];
    const int e = e_space[b * K_DIM + t];
    const float4* pr = (const float4*)(emb_r + (long)r * ED);
    const float4* pe = (const float4*)(emb_e + (long)e * ED);

    float s = 0.0f;
    #pragma unroll 8
    for (int d = 0; d < ED / 4; ++d) {
        float4 ar = pr[d];
        s = fmaf(ar.x, xs[d*4 + 0], s);
        s = fmaf(ar.y, xs[d*4 + 1], s);
        s = fmaf(ar.z, xs[d*4 + 2], s);
        s = fmaf(ar.w, xs[d*4 + 3], s);
    }
    #pragma unroll 8
    for (int d = 0; d < ED / 4; ++d) {
        float4 ae = pe[d];
        s = fmaf(ae.x, xs[256 + d*4 + 0], s);
        s = fmaf(ae.y, xs[256 + d*4 + 1], s);
        s = fmaf(ae.z, xs[256 + d*4 + 2], s);
        s = fmaf(ae.w, xs[256 + d*4 + 3], s);
    }
    s -= (1.0f - mask[b * K_DIM + t]) * HUGE_F;

    float wm = wave_max(s);
    if (lane == 0) red[wid] = wm;
    __syncthreads();
    float mx = red[0];
    #pragma unroll
    for (int i = 1; i < 8; ++i) mx = fmaxf(mx, red[i]);
    __syncthreads();

    const float ex = __expf(s - mx);

    float ws = wave_sum(ex);
    if (lane == 0) red[wid] = ws;
    __syncthreads();
    float denom = red[0];
    #pragma unroll
    for (int i = 1; i < 8; ++i) denom += red[i];
    __syncthreads();

    const float p = ex / denom;
    dist[b * K_DIM + t] = p;

    float h  = -p * __logf(p + 1e-20f);
    float wh = wave_sum(h);
    if (lane == 0) red[wid] = wh;
    __syncthreads();
    if (t == 0) {
        float hs = red[0];
        #pragma unroll
        for (int i = 1; i < 8; ++i) hs += red[i];
        entropy[b] = hs;
    }
}

// ---------------------------------------------------------------------------
// Launch
// ---------------------------------------------------------------------------
extern "C" void kernel_launch(void* const* d_in, const int* in_sizes, int n_in,
                              void* d_out, int out_size, void* d_ws, size_t ws_size,
                              hipStream_t stream) {
    const float* E     = (const float*)d_in[0];
    const float* H     = (const float*)d_in[1];
    const float* Q     = (const float*)d_in[2];
    const float* W1    = (const float*)d_in[3];
    const float* b1    = (const float*)d_in[4];
    const float* W2    = (const float*)d_in[5];
    const float* b2    = (const float*)d_in[6];
    const float* emb_r = (const float*)d_in[7];
    const float* emb_e = (const float*)d_in[8];
    const float* amask = (const float*)d_in[9];
    const int*   r_sp  = (const int*)d_in[10];
    const int*   e_sp  = (const int*)d_in[11];

    // workspace layout (256B-aligned slabs)
    char* ws = (char*)d_ws;
    size_t off = 0;
    __bf16* W1bf = (__bf16*)(ws + off); off += (size_t)AD * IN_DIM * 2;     // 768 KiB
    __bf16* W2bf = (__bf16*)(ws + off); off += (size_t)AD * AD * 2;         // 512 KiB
    __bf16* Xin  = (__bf16*)(ws + off); off += (size_t)B_DIM * IN_DIM * 2;  // 3 MiB
    __bf16* Xbf  = (__bf16*)(ws + off); off += (size_t)B_DIM * AD * 2;      // 2 MiB
    float*  X2   = (float*) (ws + off);                                     // 4 MiB

    float* dist = (float*)d_out;
    float* entr = (float*)d_out + (long)B_DIM * K_DIM;

    const int nW1 = AD * IN_DIM;   // 393216
    const int nW2 = AD * AD;       // 262144
    cvt_f32_bf16_kernel<<<(nW1 + 255) / 256, 256, 0, stream>>>(W1, W1bf, nW1);
    cvt_f32_bf16_kernel<<<(nW2 + 255) / 256, 256, 0, stream>>>(W2, W2bf, nW2);

    pack_xin_kernel<<<(B_DIM * IN_DIM / 4 + 255) / 256, 256, 0, stream>>>(E, H, Q, Xin);

    // 128 blocks x 8 waves; block -> 16x512 output strip
    gemm_wmma_kernel<IN_DIM, true ><<<128, 256, 0, stream>>>(Xin, W1bf, b1, Xbf, nullptr);
    gemm_wmma_kernel<AD,     false><<<128, 256, 0, stream>>>(Xbf, W2bf, b2, nullptr, X2);

    scores_kernel<<<B_DIM, 256, 0, stream>>>(X2, emb_r, emb_e, amask,
                                             r_sp, e_sp, dist, entr);
}
